// CausalAttention_21620865368674
// MI455X (gfx1250) — matrix operand order repaired and checked by reading, not compiled, not run on hardware
//
#include <hip/hip_runtime.h>
#include <hip/hip_bf16.h>

#define BB 4
#define TT 2048
#define DM 1024
#define NH 16
#define HD 64
#define BT (BB * TT)

typedef __attribute__((ext_vector_type(16))) __bf16 bf16x16;
typedef __attribute__((ext_vector_type(8)))  float  f32x8;

union Frag16 { bf16x16 bf; uint4 q[2]; };
union US8    { uint4 v; unsigned short e[8]; };

__device__ __forceinline__ unsigned short f2bf(float f) {
  unsigned u = __float_as_uint(f);
  unsigned r = u + 0x7FFFu + ((u >> 16) & 1u);   // round-to-nearest-even
  return (unsigned short)(r >> 16);
}

// ---- packed fp32->bf16 convert: v_cvt_pk_bf16_f32 with manual fallback ----
#if __has_builtin(__builtin_amdgcn_cvt_pk_bf16_f32)
typedef __attribute__((ext_vector_type(2))) __bf16 bf16x2;
__device__ __forceinline__ unsigned pkbf(float a, float b) {
  bf16x2 v = __builtin_amdgcn_cvt_pk_bf16_f32(a, b);
  return __builtin_bit_cast(unsigned, v);
}
#else
__device__ __forceinline__ unsigned pkbf(float a, float b) {
  return (unsigned)f2bf(a) | ((unsigned)f2bf(b) << 16);
}
#endif

// ---- async global->LDS copy (CDNA5 ASYNCcnt path), with fallback ----------
// Probe-learned signature: (AS1 int4*, AS3 int4*, imm offset, imm cpol)
typedef int gvi4 __attribute__((vector_size(16)));
typedef __attribute__((address_space(1))) gvi4 as1_i4;
typedef __attribute__((address_space(3))) gvi4 as3_i4;

#if __has_builtin(__builtin_amdgcn_global_load_async_to_lds_b128)
#define ASYNC_CP_OK 1
#define ASYNC_B128(gptr, lptr, off)                                           \
  __builtin_amdgcn_global_load_async_to_lds_b128(                             \
      (as1_i4*)(gptr), (as3_i4*)(lptr), (off), 0)
#else
#define ASYNC_CP_OK 0
#endif

__device__ __forceinline__ void wait_async0() {
#if __has_builtin(__builtin_amdgcn_s_wait_asynccnt)
  __builtin_amdgcn_s_wait_asynccnt(0);
#else
  asm volatile("s_wait_asynccnt 0" ::: "memory");
#endif
}

// ---- 16-lane row reductions: v_permlane16_b32 (VALU) with shfl fallback ----
#if __has_builtin(__builtin_amdgcn_permlane16)
__device__ __forceinline__ float pl16(float v, unsigned lo, unsigned hi) {
  unsigned u = __float_as_uint(v);
  return __uint_as_float(__builtin_amdgcn_permlane16(u, u, lo, hi, false, false));
}
__device__ __forceinline__ float rmax16(float v) {
  v = fmaxf(v, pl16(v, 0x67452301u, 0xEFCDAB89u));  // xor 1
  v = fmaxf(v, pl16(v, 0x54761032u, 0xDCFE98BAu));  // xor 2
  v = fmaxf(v, pl16(v, 0x32107654u, 0xBA98FEDCu));  // xor 4
  v = fmaxf(v, pl16(v, 0xFEDCBA98u, 0x76543210u));  // xor 8
  return v;
}
__device__ __forceinline__ float rsum16(float v) {
  v += pl16(v, 0x67452301u, 0xEFCDAB89u);
  v += pl16(v, 0x54761032u, 0xDCFE98BAu);
  v += pl16(v, 0x32107654u, 0xBA98FEDCu);
  v += pl16(v, 0xFEDCBA98u, 0x76543210u);
  return v;
}
#else
__device__ __forceinline__ float rmax16(float v) {
  v = fmaxf(v, __shfl_xor(v, 1, 32));
  v = fmaxf(v, __shfl_xor(v, 2, 32));
  v = fmaxf(v, __shfl_xor(v, 4, 32));
  v = fmaxf(v, __shfl_xor(v, 8, 32));
  return v;
}
__device__ __forceinline__ float rsum16(float v) {
  v += __shfl_xor(v, 1, 32);
  v += __shfl_xor(v, 2, 32);
  v += __shfl_xor(v, 4, 32);
  v += __shfl_xor(v, 8, 32);
  return v;
}
#endif

// ---------------------------------------------------------------------------
// bf16-WMMA GEMM: C[M,N] = A[M,K] * B[K,N] + bias.  Block tile 128x128,
// K-step 32, 8 waves = 2(M) x 4(N); each wave: 4 M-frags x 2 N-frags.
// Software pipeline: global loads for tile k+1 issue under tile-k WMMAs
// (register prefetch); bf16-A path stages via async LDS DMA.
// EPI==0: fp32 out.  EPI==1: scatter bf16 into Q/K/V head-major.
// ---------------------------------------------------------------------------
template<int EPI, typename TA>
__global__ __launch_bounds__(256) void gemm_bf16_wmma(
    const TA* __restrict__ A, const float* __restrict__ Bw,
    const float* __restrict__ bias, float* __restrict__ Cout,
    unsigned short* __restrict__ Qb, unsigned short* __restrict__ Kb,
    unsigned short* __restrict__ Vb, int M, int N, int K)
{
  __shared__ unsigned short As[128][40];  // [m][k], +8 pad
  __shared__ unsigned short Bs[128][40];  // transposed: [n][k], +8 pad

  const int tid  = threadIdx.x;
  const int lane = tid & 31;
  const int wid  = tid >> 5;
  const int wn   = (wid & 3) * 32;        // wave N offset (2 frags)
  const int wm   = (wid >> 2) * 64;       // wave M offset (4 frags)
  const int m0   = blockIdx.y * 128;
  const int n0   = blockIdx.x * 128;
  const int ln   = lane & 15;
  const int hi   = (lane >= 16);

  f32x8 acc[8];
  #pragma unroll
  for (int f = 0; f < 8; ++f)
    #pragma unroll
    for (int r = 0; r < 8; ++r) acc[f][r] = 0.0f;

  const int ar  = tid >> 1, ac  = (tid & 1) * 16;   // A: 16 elems/thread
  const int brw = tid >> 4, bcc = (tid & 15) * 8;   // B: 2 rows x 8 cols/thread

  // ---- prefetch registers ----
  float4 apf[4];      // A tile (fp32 path)
  uint4  apf2[2];     // A tile (bf16 path without async DMA)
  float4 bpf[2][2];   // B tile

  auto ld_b = [&](int k0) {
    #pragma unroll
    for (int rr = 0; rr < 2; ++rr) {
      const float4* bp =
          (const float4*)(Bw + (size_t)(k0 + brw + rr * 16) * N + n0 + bcc);
      bpf[rr][0] = bp[0];
      bpf[rr][1] = bp[1];
    }
  };
  auto ld_a = [&](int k0) {
    if constexpr (sizeof(TA) == 4) {
      const float4* ap = (const float4*)(A + (size_t)(m0 + ar) * K + k0 + ac);
      #pragma unroll
      for (int i = 0; i < 4; ++i) apf[i] = ap[i];
    }
#if !ASYNC_CP_OK
    else {
      const uint4* ap = (const uint4*)(A + (size_t)(m0 + ar) * K + k0 + ac);
      apf2[0] = ap[0];
      apf2[1] = ap[1];
    }
#endif
  };

  ld_a(0);
  ld_b(0);

  for (int k0 = 0; k0 < K; k0 += 32) {
    __syncthreads();                       // LDS free (readers of prev tile done)
    // ---- commit A tile (128x32 bf16) ----
    if constexpr (sizeof(TA) == 4) {
      #pragma unroll
      for (int i = 0; i < 4; ++i) {
        *(unsigned*)&As[ar][ac + 4 * i]     = pkbf(apf[i].x, apf[i].y);
        *(unsigned*)&As[ar][ac + 4 * i + 2] = pkbf(apf[i].z, apf[i].w);
      }
    } else {
#if ASYNC_CP_OK
      const TA* ap = A + (size_t)(m0 + ar) * K + k0 + ac;
      ASYNC_B128(ap, &As[ar][ac], 0);
      ASYNC_B128(ap, &As[ar][ac], 16);
#else
      *(uint4*)&As[ar][ac]     = apf2[0];
      *(uint4*)&As[ar][ac + 8] = apf2[1];
#endif
    }
    // ---- commit B tile transposed (Bs[n][k]) ----
    #pragma unroll
    for (int rr = 0; rr < 2; ++rr) {
      const int row = brw + rr * 16;
      unsigned u;
      u = pkbf(bpf[rr][0].x, bpf[rr][0].y);
      Bs[bcc + 0][row] = (unsigned short)u;
      Bs[bcc + 1][row] = (unsigned short)(u >> 16);
      u = pkbf(bpf[rr][0].z, bpf[rr][0].w);
      Bs[bcc + 2][row] = (unsigned short)u;
      Bs[bcc + 3][row] = (unsigned short)(u >> 16);
      u = pkbf(bpf[rr][1].x, bpf[rr][1].y);
      Bs[bcc + 4][row] = (unsigned short)u;
      Bs[bcc + 5][row] = (unsigned short)(u >> 16);
      u = pkbf(bpf[rr][1].z, bpf[rr][1].w);
      Bs[bcc + 6][row] = (unsigned short)u;
      Bs[bcc + 7][row] = (unsigned short)(u >> 16);
    }
    if constexpr (sizeof(TA) == 2 && ASYNC_CP_OK) wait_async0();
    __syncthreads();

    // ---- prefetch next tile (latency hidden under the WMMAs below) ----
    if (k0 + 32 < K) { ld_a(k0 + 32); ld_b(k0 + 32); }

    // A fragments: lanes 0-15 -> k {0..7,16..23}, lanes 16-31 -> {8..15,24..31}
    const int c0 = hi ? 8 : 0;
    Frag16 af[4];
    #pragma unroll
    for (int f = 0; f < 4; ++f) {
      const int m = wm + f * 16 + ln;
      af[f].q[0] = *(const uint4*)&As[m][c0];
      af[f].q[1] = *(const uint4*)&As[m][c0 + 16];
    }
    // B fragments: lanes 0-15 -> k {0..7,16..23}, lanes 16-31 -> {8..15,24..31}
    const int kc = hi ? 8 : 0;
    #pragma unroll
    for (int j = 0; j < 2; ++j) {
      Frag16 bf;
      bf.q[0] = *(const uint4*)&Bs[wn + j * 16 + ln][kc];
      bf.q[1] = *(const uint4*)&Bs[wn + j * 16 + ln][kc + 16];
      #pragma unroll
      for (int f = 0; f < 4; ++f) {
        acc[j * 4 + f] = __builtin_amdgcn_wmma_f32_16x16x32_bf16(
            false, af[f].bf, false, bf.bf, (short)0, acc[j * 4 + f],
            false, false);
      }
    }
  }

  // ---- epilogue ----
  #pragma unroll
  for (int j = 0; j < 2; ++j) {
    const int gcol = n0 + wn + j * 16 + ln;
    const float bv = bias[gcol];
    #pragma unroll
    for (int f = 0; f < 4; ++f) {
      if constexpr (EPI == 0) {
        #pragma unroll
        for (int r = 0; r < 8; ++r) {
          const int grow = m0 + wm + f * 16 + (hi ? 8 + r : r);
          Cout[(size_t)grow * N + gcol] = acc[j * 4 + f][r] + bv;
        }
      } else {
        const int which = gcol >> 10, cc = gcol & 1023;
        const int h = cc >> 6, dd = cc & 63;
        unsigned short* dst = (which == 0) ? Qb : (which == 1) ? Kb : Vb;
        #pragma unroll
        for (int r = 0; r < 8; r += 2) {            // adjacent rows -> t, t+1
          const int grow = m0 + wm + f * 16 + (hi ? 8 + r : r);
          const int bi = grow >> 11, t = grow & 2047;
          const unsigned u =
              pkbf(acc[j * 4 + f][r] + bv, acc[j * 4 + f][r + 1] + bv);
          const size_t idx = (((size_t)(bi * NH + h)) * TT + t) * HD + dd;
          dst[idx]      = (unsigned short)u;
          dst[idx + HD] = (unsigned short)(u >> 16);
        }
      }
    }
  }
}

// ---------------------------------------------------------------------------
// Flash attention: grid (T/128, B*H), 8 waves; each wave owns 16 q-rows.
// KV streamed in tiles of 32 with online softmax; bf16 WMMA, fp32 accum.
// Ping-pong LDS: async K DMA + V register prefetch for tile t+1 overlap the
// QK^T / softmax / PV of tile t.  P converts C-layout -> A-layout via a
// per-wave LDS patch (same-wave RAW covered by s_wait_dscnt).
// ---------------------------------------------------------------------------
__global__ __launch_bounds__(256) void attn_wmma(
    const unsigned short* __restrict__ Qb, const unsigned short* __restrict__ Kb,
    const unsigned short* __restrict__ Vb, unsigned short* __restrict__ Yb)
{
  __shared__ unsigned short Ks[2][32][72];   // [buf][kv][d]  row-major K tile
  __shared__ unsigned short Vs[2][64][40];   // [buf][d][kv]  transposed V tile
  __shared__ unsigned short Ps[8][16][40];   // per-wave P staging [q][kv]

  const int tid = threadIdx.x, lane = tid & 31, wq = tid >> 5;
  const int qt = blockIdx.x, bh = blockIdx.y;
  const int bi = bh >> 4, h = bh & 15;
  const int qbase = qt * 128;
  const size_t hoff = (size_t)bh * TT * HD;
  const unsigned short* Qh = Qb + hoff;
  const unsigned short* Kh = Kb + hoff;
  const unsigned short* Vh = Vb + hoff;

  const int ln = lane & 15;
  const int hi = (lane >= 16);
  const int q0 = qbase + wq * 16;

  // Persistent Q A-fragments for d 0..31 and 32..63
  Frag16 qf[2];
  #pragma unroll
  for (int w = 0; w < 2; ++w) {
    const unsigned short* qp = Qh + (size_t)(q0 + ln) * HD + 32 * w + (hi ? 8 : 0);
    qf[w].q[0] = *(const uint4*)qp;
    qf[w].q[1] = *(const uint4*)(qp + 16);
  }

  f32x8 o[4];
  #pragma unroll
  for (int f = 0; f < 4; ++f)
    #pragma unroll
    for (int r = 0; r < 8; ++r) o[f][r] = 0.0f;
  float mrow[8], lrow[8];
  #pragma unroll
  for (int r = 0; r < 8; ++r) { mrow[r] = -__builtin_inff(); lrow[r] = 0.0f; }

  const int kvmax_w = q0 + 15;
  const int ntiles = (qbase + 128) >> 5;
  const int skv = tid >> 3, sdc = (tid & 7) * 8;

  // ---- pipeline prologue: fetch tile 0 ----
  US8 vpf;
  uint4 kpf;
#if ASYNC_CP_OK
  ASYNC_B128(Kh + (size_t)skv * HD + sdc, &Ks[0][skv][sdc], 0);
#else
  kpf = *(const uint4*)(Kh + (size_t)skv * HD + sdc);
#endif
  vpf.v = *(const uint4*)(Vh + (size_t)skv * HD + sdc);

  for (int it = 0; it < ntiles; ++it) {
    const int kvb = it << 5;
    const int cur = it & 1, nxt = cur ^ 1;
    // ---- commit tile `it` into buffer `cur` ----
#if ASYNC_CP_OK
    wait_async0();                          // my K DMA writes to Ks[cur] done
#else
    *(uint4*)&Ks[cur][skv][sdc] = kpf;
#endif
    #pragma unroll
    for (int j = 0; j < 8; ++j) Vs[cur][sdc + j][skv] = vpf.e[j];
    __syncthreads();                        // all waves: tile `it` visible
    // ---- launch fetch of tile it+1 into buffer `nxt` (overlaps compute) ----
    if (it + 1 < ntiles) {
      const size_t roff = (size_t)(kvb + 32 + skv) * HD + sdc;
#if ASYNC_CP_OK
      ASYNC_B128(Kh + roff, &Ks[nxt][skv][sdc], 0);
#else
      kpf = *(const uint4*)(Kh + roff);
#endif
      vpf.v = *(const uint4*)(Vh + roff);
    }
    if (kvb > kvmax_w) continue;            // wave-uniform; fully masked tile

    // ---- S = Q K^T  (16q x 32kv as two C-fragments) ----
    f32x8 s[2];
    #pragma unroll
    for (int j = 0; j < 2; ++j) {
      #pragma unroll
      for (int r = 0; r < 8; ++r) s[j][r] = 0.0f;
      #pragma unroll
      for (int w = 0; w < 2; ++w) {
        Frag16 kf;                      // B-frag of K^T: n=kv, k=d (contig rows)
        const int dch = 32 * w + (hi ? 8 : 0);
        kf.q[0] = *(const uint4*)&Ks[cur][j * 16 + ln][dch];
        kf.q[1] = *(const uint4*)&Ks[cur][j * 16 + ln][dch + 16];
        s[j] = __builtin_amdgcn_wmma_f32_16x16x32_bf16(
            false, qf[w].bf, false, kf.bf, (short)0, s[j], false, false);
      }
    }

    // ---- causal mask + online softmax (VALU permlane16 row reductions) ----
    float sc[8];
    #pragma unroll
    for (int r = 0; r < 8; ++r) {
      const int qrow = q0 + (hi ? 8 + r : r);
      float x0 = s[0][r] * 0.125f;                 // 1/sqrt(64)
      float x1 = s[1][r] * 0.125f;
      if (kvb + ln > qrow)      x0 = -__builtin_inff();
      if (kvb + 16 + ln > qrow) x1 = -__builtin_inff();
      const float mx    = rmax16(fmaxf(x0, x1));
      const float mnew  = fmaxf(mrow[r], mx);
      const float scale = __expf(mrow[r] - mnew);
      const float p0 = __expf(x0 - mnew);
      const float p1 = __expf(x1 - mnew);
      lrow[r] = lrow[r] * scale + rsum16(p0 + p1);
      mrow[r] = mnew;
      sc[r] = scale;
      const int prow = hi ? 8 + r : r;             // C-layout -> LDS row-major
      const unsigned u = pkbf(p0, p1);
      Ps[wq][prow][ln]      = (unsigned short)u;
      Ps[wq][prow][16 + ln] = (unsigned short)(u >> 16);
    }
    asm volatile("s_wait_dscnt 0" ::: "memory");   // same-wave LDS RAW

    // ---- reload P as A-fragment, O = O*scale + P V ----
    Frag16 pf;
    {
      const int c0 = hi ? 8 : 0;
      pf.q[0] = *(const uint4*)&Ps[wq][ln][c0];
      pf.q[1] = *(const uint4*)&Ps[wq][ln][c0 + 16];
    }
    #pragma unroll
    for (int f = 0; f < 4; ++f) {
      #pragma unroll
      for (int r = 0; r < 8; ++r) o[f][r] *= sc[r];
      Frag16 vf;                        // B-frag of V: n=d, k=kv (Vs transposed)
      const int kc = hi ? 8 : 0;
      vf.q[0] = *(const uint4*)&Vs[cur][f * 16 + ln][kc];
      vf.q[1] = *(const uint4*)&Vs[cur][f * 16 + ln][kc + 16];
      o[f] = __builtin_amdgcn_wmma_f32_16x16x32_bf16(
          false, pf.bf, false, vf.bf, (short)0, o[f], false, false);
    }
  }

  // ---- normalize, write Y bf16 in token-major [B*T, C] for proj GEMM ----
  #pragma unroll
  for (int r = 0; r < 8; ++r) {
    const float inv = 1.0f / lrow[r];
    const int qrow = q0 + (hi ? 8 + r : r);
    unsigned short* yp = Yb + ((size_t)(bi * TT + qrow)) * DM + h * HD + ln;
    #pragma unroll
    for (int f = 0; f < 4; f += 2) {
      const unsigned u = pkbf(o[f][r] * inv, o[f + 1][r] * inv);
      yp[f * 16]      = (unsigned short)u;
      yp[f * 16 + 16] = (unsigned short)(u >> 16);
    }
  }
}

extern "C" void kernel_launch(void* const* d_in, const int* in_sizes, int n_in,
                              void* d_out, int out_size, void* d_ws, size_t ws_size,
                              hipStream_t stream) {
  const float* x      = (const float*)d_in[0];
  const float* W_qkv  = (const float*)d_in[1];
  const float* b_qkv  = (const float*)d_in[2];
  const float* W_proj = (const float*)d_in[3];
  const float* b_proj = (const float*)d_in[4];

  const size_t QE = (size_t)BB * NH * TT * HD;   // 8 Mi elements per buffer
  unsigned short* Qb = (unsigned short*)d_ws;
  unsigned short* Kb = Qb + QE;
  unsigned short* Vb = Kb + QE;
  unsigned short* Yb = Vb + QE;                  // total 64 MB bf16

  // 1) QKV GEMM: [8192,1024] x [1024,3072] -> Q/K/V bf16 head-major
  gemm_bf16_wmma<1, float><<<dim3((3 * DM) / 128, BT / 128), 256, 0, stream>>>(
      x, W_qkv, b_qkv, nullptr, Qb, Kb, Vb, BT, 3 * DM, DM);

  // 2) Flash attention -> Y bf16 [8192,1024]
  attn_wmma<<<dim3(TT / 128, BB * NH), 256, 0, stream>>>(Qb, Kb, Vb, Yb);

  // 3) Output projection: [8192,1024] x [1024,1024] + b -> fp32 d_out
  gemm_bf16_wmma<0, unsigned short><<<dim3(DM / 128, BT / 128), 256, 0, stream>>>(
      Yb, W_proj, b_proj, (float*)d_out, nullptr, nullptr, nullptr, BT, DM, DM);
}
